// GroupedQueryAttention_28089086116364
// MI455X (gfx1250) — matrix operand order repaired and checked by reading, not compiled, not run on hardware
//
#include <hip/hip_runtime.h>
#include <hip/hip_bf16.h>
#include <cstdint>

// ---------------- problem constants ----------------
#define DM    1024          // d_model
#define NH    16            // query heads
#define NKV   4             // kv heads
#define DK    64            // head dim
#define BB    2             // batch
#define SS    2048          // seq len
#define MTOK  (BB*SS)       // 4096 tokens
#define KVDIM (NKV*DK)      // 256

// ---------------- WMMA fragment types ----------------
typedef __attribute__((ext_vector_type(16))) __bf16          v16bf;
typedef __attribute__((ext_vector_type(8)))  float           v8f;
typedef __attribute__((ext_vector_type(16))) unsigned short  v16u;
// ---------------- TDM descriptor vector types ----------------
typedef __attribute__((ext_vector_type(4))) unsigned int     u32x4;
typedef __attribute__((ext_vector_type(4))) int              i32x4;
typedef __attribute__((ext_vector_type(8))) int              i32x8;

union Frag {
    v16bf bf;
    v16u  us;
    uint4 q[2];   // two 16-byte pieces for vector loads
};

__device__ __forceinline__ unsigned short f2bf(float f) {
    unsigned u = __builtin_bit_cast(unsigned, f);
    u += 0x7FFFu + ((u >> 16) & 1u);          // round-to-nearest-even
    return (unsigned short)(u >> 16);
}

// ---------------- Tensor Data Mover: 2-D bf16 tile -> LDS ----------------
// D# per CDNA5 ISA. Groups 2/3 zero (2-D tensor).
// 6-arg builtin form (amdgpu-toolchain / clang-23): extra zero int32x8 group.
// Tracked by TENSORcnt; caller pairs with s_wait_tensorcnt + barrier.
__device__ __forceinline__ void tdm_load_2d(const unsigned short* gsrc,
                                            unsigned lds_byte_off,
                                            unsigned tensor_dim0,   // row length (elems)
                                            unsigned tensor_dim1,   // #rows
                                            unsigned tile_w,        // tile cols (elems)
                                            unsigned tile_h,        // tile rows
                                            unsigned row_stride) {  // elems between rows
#if __has_builtin(__builtin_amdgcn_tensor_load_to_lds)
    uint64_t ga = (uint64_t)(uintptr_t)gsrc;
    u32x4 g0;
    g0.x = 1u;                                           // count=1, user D#
    g0.y = lds_byte_off;                                 // lds_addr
    g0.z = (unsigned)(ga & 0xFFFFFFFFu);                 // global_addr[31:0]
    g0.w = (unsigned)((ga >> 32) & 0x1FFFFFFu)           // global_addr[56:32]
         | (2u << 30);                                   // type = 2 ("image")
    i32x8 g1;
    g1[0] = (int)(1u << 16);                             // data_size=1 -> 2 bytes
    g1[1] = (int)((tensor_dim0 & 0xFFFFu) << 16);        // tensor_dim0[15:0]
    g1[2] = (int)((tensor_dim0 >> 16)
         | ((tensor_dim1 & 0xFFFFu) << 16));             // dim0[31:16] | dim1[15:0]
    g1[3] = (int)((tensor_dim1 >> 16)
         | ((tile_w & 0xFFFFu) << 16));                  // dim1[31:16] | tile_dim0
    g1[4] = (int)(tile_h & 0xFFFFu);                     // tile_dim1 (tile_dim2=0)
    g1[5] = (int)row_stride;                             // tensor_dim0_stride[31:0]
    g1[6] = 0;                                           // stride0[47:32]|stride1[15:0]
    g1[7] = 0;                                           // stride1[47:16]
    i32x4 z4 = {0, 0, 0, 0};
    i32x8 z8 = {0, 0, 0, 0, 0, 0, 0, 0};
    __builtin_amdgcn_tensor_load_to_lds(g0, g1, z4, z4, z8, 0);
#else
    (void)gsrc; (void)lds_byte_off; (void)tensor_dim0; (void)tensor_dim1;
    (void)tile_w; (void)tile_h; (void)row_stride;
#endif
}

// s_wait_tensorcnt needs a literal immediate -> template constant.
template <int N>
__device__ __forceinline__ void tdm_wait() {
#if __has_builtin(__builtin_amdgcn_s_wait_tensorcnt)
    __builtin_amdgcn_s_wait_tensorcnt(N);
#endif
}

// ---------------- fp32 -> bf16 conversion ----------------
__global__ void __launch_bounds__(256)
cvt_f32_bf16(const float* __restrict__ src,
             unsigned short* __restrict__ dst, int n) {
    for (int i = blockIdx.x * blockDim.x + threadIdx.x; i < n;
         i += gridDim.x * blockDim.x)
        dst[i] = f2bf(src[i]);
}

// ---------------- bf16 WMMA GEMM: C[M,N] = A[M,K] @ W[K,N] ----------------
// block = 256 threads (8 waves), block tile 64x64, wave tile 16x32.
// W tiles (32x64 bf16, 4 KB) are double-buffered in LDS via TDM; wave 0 issues
// tensor_load_to_lds for tile k+1 while all waves compute tile k.
// grid = (N/64, M/64); dynamic LDS = 8192 bytes.
template <bool F32OUT>
__global__ void __launch_bounds__(256, 1)
gemm_bf16(const unsigned short* __restrict__ A,
          const unsigned short* __restrict__ W,
          void* __restrict__ Cout,
          int N, int K, float scale) {
    extern __shared__ unsigned short ldsW[];   // 2 x (32 x 64) bf16
    const int lane = threadIdx.x & 31;
    const int wave = threadIdx.x >> 5;
    const int half = lane >> 4;
    const int ln   = lane & 15;
    const int wm = wave & 3, wn = wave >> 2;
    const int bm  = blockIdx.y * 64 + wm * 16;
    const int bn0 = blockIdx.x * 64;
    const int bn  = bn0 + wn * 32;

    v8f c0 = {0.f,0.f,0.f,0.f,0.f,0.f,0.f,0.f};
    v8f c1 = c0;
    const unsigned short* arow = A + (size_t)(bm + ln) * K;

    if (wave == 0)                                     // prime the pipeline
        tdm_load_2d(W + bn0, 0, (unsigned)N, (unsigned)K, 64, 32, (unsigned)N);

    int buf = 0;
    for (int kk = 0; kk < K; kk += 32) {
        if (wave == 0) {
            if (kk + 32 < K) {                         // prefetch next W tile
                tdm_load_2d(W + (size_t)(kk + 32) * N + bn0,
                            (unsigned)((buf ^ 1) * 4096),
                            (unsigned)N, (unsigned)K, 64, 32, (unsigned)N);
                tdm_wait<1>();                         // current tile ready
            } else {
                tdm_wait<0>();                         // last tile ready
            }
        }
        __syncthreads();

        Frag a, b0, b1;
        __builtin_prefetch(arow + kk + 128, 0, 1);     // global_prefetch_b8
        // A fragment: row = lane%16; element e -> K=half*16+e
        a.q[0] = *(const uint4*)(arow + kk + half * 16);
        a.q[1] = *(const uint4*)(arow + kk + half * 16 + 8);
        // B fragment from LDS tile: col = lane%16; element e -> k = half*16+e
        const unsigned short* wt = ldsW + buf * 2048;
        #pragma unroll
        for (int e = 0; e < 16; ++e) {
            const unsigned short* wr = wt + (half * 16 + e) * 64 + wn * 32 + ln;
            b0.us[e] = wr[0];
            b1.us[e] = wr[16];
        }
        c0 = __builtin_amdgcn_wmma_f32_16x16x32_bf16(false, a.bf, false, b0.bf,
                                                     (short)0, c0, false, false);
        c1 = __builtin_amdgcn_wmma_f32_16x16x32_bf16(false, a.bf, false, b1.bf,
                                                     (short)0, c1, false, false);
        __syncthreads();                               // done reading buf
        buf ^= 1;
    }
    // C/D layout: VGPR r -> row r + half*8, lane%16 -> col
    #pragma unroll
    for (int r = 0; r < 8; ++r) {
        int row = bm + half * 8 + r;
        if constexpr (F32OUT) {
            float* C = (float*)Cout;
            C[(size_t)row * N + bn + ln]      = c0[r] * scale;
            C[(size_t)row * N + bn + 16 + ln] = c1[r] * scale;
        } else {
            unsigned short* C = (unsigned short*)Cout;
            C[(size_t)row * N + bn + ln]      = f2bf(c0[r] * scale);
            C[(size_t)row * N + bn + 16 + ln] = f2bf(c1[r] * scale);
        }
    }
}

// ---------------- flash attention (GQA) ----------------
// block = 128 threads (4 waves); block = (b, h, 64 q-rows); wave = 16 q-rows.
// K/V 32x64 bf16 tiles are DOUBLE-BUFFERED in LDS via TDM: wave 0 issues the
// next pair of tensor_load_to_lds before compute of the current pair, so the
// tensor DMA overlaps WMMA + softmax. All 4 waves share the tiles.
// grid = (S/64, NH, B); dynamic LDS = 20480 bytes:
//   [0,8K)  K tiles x2   [8K,16K) V tiles x2   [16K,20K) per-wave P tiles
__global__ void __launch_bounds__(128, 1)
flash_attn(const unsigned short* __restrict__ Qp,
           const unsigned short* __restrict__ Kp,
           const unsigned short* __restrict__ Vp,
           unsigned short* __restrict__ Ao) {
    extern __shared__ unsigned short fsh[];
    unsigned short* ldsKb = fsh;            // 2 x (32 keys x 64 d)
    unsigned short* ldsVb = fsh + 4096;     // 2 x (32 keys x 64 d)
    unsigned short* plds  = fsh + 8192;     // 4 waves x (16 x 32)

    const int lane = threadIdx.x & 31;
    const int wave = threadIdx.x >> 5;
    const int half = lane >> 4;
    const int ln   = lane & 15;
    const int b  = blockIdx.z;
    const int h  = blockIdx.y;
    const int qt = blockIdx.x;
    const int kvh = h >> 2;                // 4 query heads per kv head
    const int q0  = qt * 64 + wave * 16;

    // Q fragments for the two d-chunks (K=32 each), loaded once
    Frag aq[2];
    const unsigned short* qrow = Qp + (size_t)(b * SS + q0 + ln) * DM + h * DK;
    #pragma unroll
    for (int c = 0; c < 2; ++c) {
        aq[c].q[0] = *(const uint4*)(qrow + c * 32 + half * 16);
        aq[c].q[1] = *(const uint4*)(qrow + c * 32 + half * 16 + 8);
    }

    v8f oacc[4];
    #pragma unroll
    for (int j = 0; j < 4; ++j) oacc[j] = (v8f){0.f,0.f,0.f,0.f,0.f,0.f,0.f,0.f};
    float m[8], l[8];
    #pragma unroll
    for (int i = 0; i < 8; ++i) { m[i] = -3.0e38f; l[i] = 0.f; }

    if (wave == 0) {                      // prime first K/V tiles -> buffer 0
        tdm_load_2d(Kp + (size_t)(b * SS) * KVDIM + kvh * DK, 0,
                    KVDIM, MTOK, 64, 32, KVDIM);
        tdm_load_2d(Vp + (size_t)(b * SS) * KVDIM + kvh * DK, 8192,
                    KVDIM, MTOK, 64, 32, KVDIM);
    }

    int buf = 0;
    for (int kb = 0; kb < SS; kb += 32) {
        if (wave == 0) {
            if (kb + 32 < SS) {           // DMA next K/V pair into other buffer
                tdm_load_2d(Kp + (size_t)(b * SS + kb + 32) * KVDIM + kvh * DK,
                            (unsigned)((buf ^ 1) * 4096),
                            KVDIM, MTOK, 64, 32, KVDIM);
                tdm_load_2d(Vp + (size_t)(b * SS + kb + 32) * KVDIM + kvh * DK,
                            (unsigned)(8192 + (buf ^ 1) * 4096),
                            KVDIM, MTOK, 64, 32, KVDIM);
                tdm_wait<2>();            // current pair retired (in-order)
            } else {
                tdm_wait<0>();            // last pair retired
            }
        }
        __syncthreads();
        const unsigned short* ldsK = ldsKb + buf * 2048;
        const unsigned short* ldsV = ldsVb + buf * 2048;

        // ---- S = Q @ K^T : B-frag col = key, contraction = d (contiguous) ----
        v8f s0, s1;
        {
            v8f z = {0.f,0.f,0.f,0.f,0.f,0.f,0.f,0.f};
            #pragma unroll
            for (int c = 0; c < 2; ++c) {
                Frag bk0, bk1;             // d-chunks 0 and 1 for this col group
                const unsigned short* kp = ldsK + (c * 16 + ln) * 64;
                bk0.q[0] = *(const uint4*)(kp + half * 16);
                bk0.q[1] = *(const uint4*)(kp + half * 16 + 8);
                bk1.q[0] = *(const uint4*)(kp + 32 + half * 16);
                bk1.q[1] = *(const uint4*)(kp + 32 + half * 16 + 8);
                v8f s = __builtin_amdgcn_wmma_f32_16x16x32_bf16(
                    false, aq[0].bf, false, bk0.bf, (short)0, z, false, false);
                s = __builtin_amdgcn_wmma_f32_16x16x32_bf16(
                    false, aq[1].bf, false, bk1.bf, (short)0, s, false, false);
                if (c == 0) s0 = s; else s1 = s;
            }
        }

        // ---- online softmax; row i+half*8 lives in the 16 lanes of a half ----
        #pragma unroll
        for (int i = 0; i < 8; ++i) {
            float t = fmaxf(s0[i], s1[i]);
            t = fmaxf(t, __shfl_xor(t, 1, 32));
            t = fmaxf(t, __shfl_xor(t, 2, 32));
            t = fmaxf(t, __shfl_xor(t, 4, 32));
            t = fmaxf(t, __shfl_xor(t, 8, 32));
            float mn   = fmaxf(m[i], t);
            float corr = __expf(m[i] - mn);
            float e0   = __expf(s0[i] - mn);
            float e1   = __expf(s1[i] - mn);
            float rs   = e0 + e1;
            rs += __shfl_xor(rs, 1, 32);
            rs += __shfl_xor(rs, 2, 32);
            rs += __shfl_xor(rs, 4, 32);
            rs += __shfl_xor(rs, 8, 32);
            l[i] = l[i] * corr + rs;
            m[i] = mn;
            #pragma unroll
            for (int j = 0; j < 4; ++j) oacc[j][i] *= corr;
            int row = half * 8 + i;        // stage P (bf16) in this wave's tile
            plds[wave * 512 + row * 32 + ln]      = f2bf(e0);
            plds[wave * 512 + row * 32 + 16 + ln] = f2bf(e1);
        }

        // ---- reload P in A-fragment layout (in-wave LDS is in-order) ----
        Frag p;
        const unsigned short* lp = plds + wave * 512 + ln * 32;
        p.q[0] = *(const uint4*)(lp + half * 16);
        p.q[1] = *(const uint4*)(lp + half * 16 + 8);

        // ---- O += P @ V ; contraction = key, col = d ----
        #pragma unroll
        for (int j = 0; j < 4; ++j) {
            Frag bv;
            #pragma unroll
            for (int e = 0; e < 16; ++e)
                bv.us[e] = ldsV[(half * 16 + e) * 64 + j * 16 + ln];
            oacc[j] = __builtin_amdgcn_wmma_f32_16x16x32_bf16(
                false, p.bf, false, bv.bf, (short)0, oacc[j], false, false);
        }

        __syncthreads();                   // everyone done with this buffer
        buf ^= 1;
    }

    // normalize and write bf16 attention output in [token, h*DK+d] layout
    #pragma unroll
    for (int i = 0; i < 8; ++i) {
        float inv = 1.0f / l[i];
        unsigned short* op =
            Ao + (size_t)(b * SS + q0 + half * 8 + i) * DM + h * DK;
        #pragma unroll
        for (int j = 0; j < 4; ++j)
            op[j * 16 + ln] = f2bf(oacc[j][i] * inv);
    }
}

// ---------------- host launcher ----------------
extern "C" void kernel_launch(void* const* d_in, const int* in_sizes, int n_in,
                              void* d_out, int out_size, void* d_ws, size_t ws_size,
                              hipStream_t stream) {
    (void)in_sizes; (void)n_in; (void)out_size; (void)ws_size;
    const float* q  = (const float*)d_in[0];
    const float* k  = (const float*)d_in[1];
    const float* v  = (const float*)d_in[2];
    const float* Wq = (const float*)d_in[3];
    const float* Wk = (const float*)d_in[4];
    const float* Wv = (const float*)d_in[5];
    const float* Wo = (const float*)d_in[6];

    unsigned short* ws = (unsigned short*)d_ws;
    size_t off = 0;
    auto take = [&](size_t n) { unsigned short* p = ws + off; off += n; return p; };
    unsigned short* qb  = take((size_t)MTOK * DM);     // bf16 inputs
    unsigned short* kb  = take((size_t)MTOK * DM);
    unsigned short* vb  = take((size_t)MTOK * DM);
    unsigned short* wqb = take((size_t)DM * DM);       // bf16 weights
    unsigned short* wkb = take((size_t)DM * KVDIM);
    unsigned short* wvb = take((size_t)DM * KVDIM);
    unsigned short* wob = take((size_t)DM * DM);
    unsigned short* Qp  = take((size_t)MTOK * DM);     // projected (Q pre-scaled)
    unsigned short* Kp  = take((size_t)MTOK * KVDIM);
    unsigned short* Vp  = take((size_t)MTOK * KVDIM);
    unsigned short* Ao  = take((size_t)MTOK * DM);     // attention output

    // 1) fp32 -> bf16 conversions
    cvt_f32_bf16<<<2048, 256, 0, stream>>>(q,  qb,  MTOK * DM);
    cvt_f32_bf16<<<2048, 256, 0, stream>>>(k,  kb,  MTOK * DM);
    cvt_f32_bf16<<<2048, 256, 0, stream>>>(v,  vb,  MTOK * DM);
    cvt_f32_bf16<<<1024, 256, 0, stream>>>(Wq, wqb, DM * DM);
    cvt_f32_bf16<<<256,  256, 0, stream>>>(Wk, wkb, DM * KVDIM);
    cvt_f32_bf16<<<256,  256, 0, stream>>>(Wv, wvb, DM * KVDIM);
    cvt_f32_bf16<<<1024, 256, 0, stream>>>(Wo, wob, DM * DM);

    // 2) projections (Q scaled by 1/sqrt(DK) = 0.125)
    dim3 gq(DM / 64,    MTOK / 64);
    dim3 gk(KVDIM / 64, MTOK / 64);
    gemm_bf16<false><<<gq, 256, 8192, stream>>>(qb, wqb, Qp, DM,    DM, 0.125f);
    gemm_bf16<false><<<gk, 256, 8192, stream>>>(kb, wkb, Kp, KVDIM, DM, 1.0f);
    gemm_bf16<false><<<gk, 256, 8192, stream>>>(vb, wvb, Vp, KVDIM, DM, 1.0f);

    // 3) flash attention (double-buffered TDM K/V tiles in LDS)
    flash_attn<<<dim3(SS / 64, NH, BB), 128, 20480, stream>>>(Qp, Kp, Vp, Ao);

    // 4) output projection -> fp32 d_out
    gemm_bf16<true><<<gq, 256, 8192, stream>>>(Ao, wob, d_out, DM, DM, 1.0f);
}
